// DialogueRNNCell_57183194579286
// MI455X (gfx1250) — matrix operand order repaired and checked by reading, not compiled, not run on hardware
//
#include <hip/hip_runtime.h>
#include <hip/hip_bf16.h>
#include <stdint.h>

typedef __attribute__((ext_vector_type(16))) __bf16       v16bf;
typedef __attribute__((ext_vector_type(8)))  float        v8f;
typedef __attribute__((ext_vector_type(4)))  unsigned int uint4v;

namespace {
constexpr int B  = 2048;
constexpr int P  = 2;
constexpr int T  = 128;
constexpr int H  = 512;     // U = G = DP = E
constexpr int N3 = 3 * H;   // 1536 gate columns
constexpr int K1 = 2 * H;   // 1024 concat input dim
}

__device__ __forceinline__ unsigned short f2bf(float f) {
  unsigned int u = __float_as_uint(f);
  u += 0x7FFFu + ((u >> 16) & 1u);           // round-to-nearest-even
  return (unsigned short)(u >> 16);
}
__device__ __forceinline__ float sigmoidf(float x) { return 1.f / (1.f + __expf(-x)); }

// ---------------------------------------------------------------- conversions
__global__ void k_f2bf(const float* __restrict__ s, unsigned short* __restrict__ d, int n) {
  int i = blockIdx.x * blockDim.x + threadIdx.x;
  if (i < n) d[i] = f2bf(s[i]);
}

// xg = [utterance | last_personal_state[b, spk]]  (B x 1024), bf16
__global__ void k_build_xg(const float* __restrict__ utt, const float* __restrict__ lps,
                           const float* __restrict__ pm, unsigned short* __restrict__ xg) {
  int i = blockIdx.x * blockDim.x + threadIdx.x;
  if (i >= B * K1) return;
  int b = i >> 10, c = i & (K1 - 1);
  float v;
  if (c < H) {
    v = utt[(size_t)b * H + c];
  } else {
    int spk = pm[b * 2 + 1] > pm[b * 2 + 0] ? 1 : 0;
    v = lps[((size_t)b * P + spk) * H + (c - H)];
  }
  xg[i] = f2bf(v);
}

// uss = [utterance | cur_speaker[b, spk]]  (B x 1024), bf16
__global__ void k_build_uss(const float* __restrict__ utt, const float* __restrict__ cs,
                            const float* __restrict__ pm, unsigned short* __restrict__ uss) {
  int i = blockIdx.x * blockDim.x + threadIdx.x;
  if (i >= B * K1) return;
  int b = i >> 10, c = i & (K1 - 1);
  float v;
  if (c < H) {
    v = utt[(size_t)b * H + c];
  } else {
    int spk = pm[b * 2 + 1] > pm[b * 2 + 0] ? 1 : 0;
    v = cs[((size_t)b * P + spk) * H + (c - H)];
  }
  uss[i] = f2bf(v);
}

// utterance -> cols [0,512) of uc (B x 1024 bf16)
__global__ void k_concat_utt(const float* __restrict__ utt, unsigned short* __restrict__ dst) {
  int i = blockIdx.x * blockDim.x + threadIdx.x;
  if (i >= B * H) return;
  int b = i >> 9, c = i & (H - 1);
  dst[(size_t)b * K1 + c] = f2bf(utt[i]);
}

// ---------------------------------------------------------------- attention
// scale[t*B+b] = dot(global_hist[t,b,:], attn_w)   (one wave per row)
__global__ void k_scale(const float* __restrict__ gh, const float* __restrict__ aw,
                        float* __restrict__ scale) {
  int lane = threadIdx.x & 31, wave = threadIdx.x >> 5;
  int row  = blockIdx.x * 8 + wave;              // row in [0, T*B)
  const float* p = gh + (size_t)row * H;
  float acc = 0.f;
#pragma unroll
  for (int i = 0; i < H / 32; ++i) acc += p[lane + i * 32] * aw[lane + i * 32];
#pragma unroll
  for (int off = 16; off > 0; off >>= 1) acc += __shfl_xor(acc, off, 32);
  if (lane == 0) scale[row] = acc;
}

// softmax over t (128 values per b); also emits attn_weight[b,0,t]
__global__ void k_softmax(const float* __restrict__ scale, float* __restrict__ alpha,
                          float* __restrict__ out_attn) {
  __shared__ float red[T];
  int b = blockIdx.x, t = threadIdx.x;
  float v = scale[t * B + b];
  red[t] = v; __syncthreads();
  for (int s = T / 2; s > 0; s >>= 1) { if (t < s) red[t] = fmaxf(red[t], red[t + s]); __syncthreads(); }
  float mx = red[0]; __syncthreads();
  float e = __expf(v - mx);
  red[t] = e; __syncthreads();
  for (int s = T / 2; s > 0; s >>= 1) { if (t < s) red[t] += red[t + s]; __syncthreads(); }
  float a = e / red[0];
  alpha[t * B + b]    = a;
  out_attn[b * T + t] = a;
}

// pooled[b,g] = sum_t alpha[t,b]*gh[t,b,g]  -> uc cols [512,1024) as bf16
__global__ void k_pooled(const float* __restrict__ alpha, const float* __restrict__ gh,
                         unsigned short* __restrict__ uc) {
  int i = blockIdx.x * blockDim.x + threadIdx.x;
  if (i >= B * H) return;
  int b = i >> 9, g = i & (H - 1);
  float acc = 0.f;
  for (int t = 0; t < T; ++t)
    acc += alpha[t * B + b] * gh[((size_t)t * B + b) * H + g];
  uc[(size_t)b * K1 + H + g] = f2bf(acc);
}

// ---------------------------------------------------------------- WMMA GEMM
// A fragment: 16x32 bf16 tile. Lane L holds row M = L&15.
// Lanes 0-15:  K = k0+{0..7, 16..23}; lanes 16-31: K = k0+{8..15, 24..31}.
__device__ __forceinline__ v16bf load_a_frag(const unsigned short* __restrict__ A, int ld,
                                             int row0, int k0, int lane) {
  const unsigned short* q = A + (size_t)(row0 + (lane & 15)) * ld + k0 + ((lane >> 4) << 3);
  union { v16bf v; uint4v u[2]; } x;
  x.u[0] = *(const uint4v*)q;
  x.u[1] = *(const uint4v*)(q + 16);
  return x.v;
}
// B fragment: 32x16 bf16 tile, column n = lane&15 comes from W row n (W is N x K).
// Lanes 0-15 hold K = k0+{0..7, 16..23}, lanes 16-31 hold K = k0+{8..15, 24..31} (matches A).
__device__ __forceinline__ v16bf load_b_frag(const unsigned short* __restrict__ W, int ld,
                                             int col0, int k0, int lane) {
  const unsigned short* q = W + (size_t)(col0 + (lane & 15)) * ld + k0 + ((lane >> 4) << 3);
  union { v16bf v; uint4v u[2]; } x;
  x.u[0] = *(const uint4v*)q;
  x.u[1] = *(const uint4v*)(q + 16);
  return x.v;
}

// C[M,N] = A[M,K](bf16) * W[N,K]^T(bf16) + bias[N], fp32 accumulate.
// Block = 256 threads = 8 waves; block tile 128x128; wave tile 32x64 (2x4 WMMA tiles).
__global__ __launch_bounds__(256) void k_gemm_bf16(const unsigned short* __restrict__ A,
                                                   const unsigned short* __restrict__ W,
                                                   const float* __restrict__ bias,
                                                   float* __restrict__ C,
                                                   int M, int N, int K) {
  int lane = threadIdx.x & 31, wave = threadIdx.x >> 5;
  int mBase = blockIdx.y * 128 + (wave >> 1) * 32;
  int nBase = blockIdx.x * 128 + (wave & 1) * 64;
  (void)M;

  v8f z = {0.f, 0.f, 0.f, 0.f, 0.f, 0.f, 0.f, 0.f};
  v8f acc[2][4];
#pragma unroll
  for (int i = 0; i < 2; ++i)
#pragma unroll
    for (int j = 0; j < 4; ++j) acc[i][j] = z;

  for (int k0 = 0; k0 < K; k0 += 32) {
    if (k0 + 32 < K) {                 // hint next K-tile into cache (global_prefetch_b8)
      __builtin_prefetch(A + (size_t)(mBase + (lane & 15)) * K + k0 + 32, 0, 1);
      __builtin_prefetch(W + (size_t)(nBase + lane) * K + k0 + 32, 0, 1);
    }
    v16bf a0 = load_a_frag(A, K, mBase,      k0, lane);
    v16bf a1 = load_a_frag(A, K, mBase + 16, k0, lane);
    v16bf b0 = load_b_frag(W, K, nBase +  0, k0, lane);
    v16bf b1 = load_b_frag(W, K, nBase + 16, k0, lane);
    v16bf b2 = load_b_frag(W, K, nBase + 32, k0, lane);
    v16bf b3 = load_b_frag(W, K, nBase + 48, k0, lane);

    acc[0][0] = __builtin_amdgcn_wmma_f32_16x16x32_bf16(false, a0, false, b0, (short)0, acc[0][0], false, false);
    acc[0][1] = __builtin_amdgcn_wmma_f32_16x16x32_bf16(false, a0, false, b1, (short)0, acc[0][1], false, false);
    acc[0][2] = __builtin_amdgcn_wmma_f32_16x16x32_bf16(false, a0, false, b2, (short)0, acc[0][2], false, false);
    acc[0][3] = __builtin_amdgcn_wmma_f32_16x16x32_bf16(false, a0, false, b3, (short)0, acc[0][3], false, false);
    acc[1][0] = __builtin_amdgcn_wmma_f32_16x16x32_bf16(false, a1, false, b0, (short)0, acc[1][0], false, false);
    acc[1][1] = __builtin_amdgcn_wmma_f32_16x16x32_bf16(false, a1, false, b1, (short)0, acc[1][1], false, false);
    acc[1][2] = __builtin_amdgcn_wmma_f32_16x16x32_bf16(false, a1, false, b2, (short)0, acc[1][2], false, false);
    acc[1][3] = __builtin_amdgcn_wmma_f32_16x16x32_bf16(false, a1, false, b3, (short)0, acc[1][3], false, false);
  }

  // C/D layout: lanes 0-15: N=lane, M=e; lanes 16-31: N=lane-16, M=8+e.
  int mo = (lane >> 4) << 3;
  int nc = lane & 15;
#pragma unroll
  for (int i = 0; i < 2; ++i)
#pragma unroll
    for (int j = 0; j < 4; ++j) {
      int n = nBase + j * 16 + nc;
      float bb = bias[n];
#pragma unroll
      for (int e = 0; e < 8; ++e) {
        int m = mBase + i * 16 + mo + e;
        C[(size_t)m * N + n] = acc[i][j][e] + bb;
      }
    }
}

// ---------------------------------------------------------------- GRU elementwise
__global__ void k_gru_simple(const float* __restrict__ gi, const float* __restrict__ gh,
                             const float* __restrict__ h, float* __restrict__ out, int n) {
  int i = blockIdx.x * blockDim.x + threadIdx.x;
  if (i >= n) return;
  int b = i / H, j = i - b * H;
  size_t r = (size_t)b * N3;
  float rr = sigmoidf(gi[r + j] + gh[r + j]);
  float zz = sigmoidf(gi[r + H + j] + gh[r + H + j]);
  float nn = tanhf(gi[r + 2 * H + j] + rr * gh[r + 2 * H + j]);
  out[i] = (1.f - zz) * nn + zz * h[i];
}

// speaker GRU: gi broadcast over p; gh has row (b*2+p)
__global__ void k_gru_speaker(const float* __restrict__ gi, const float* __restrict__ gh,
                              const float* __restrict__ lps, float* __restrict__ cs) {
  int i = blockIdx.x * blockDim.x + threadIdx.x;
  if (i >= B * P * H) return;
  int j = i & (H - 1);
  int bp = i >> 9;
  int b = bp >> 1;
  size_t ri = (size_t)b * N3, rh = (size_t)bp * N3;
  float rr = sigmoidf(gi[ri + j] + gh[rh + j]);
  float zz = sigmoidf(gi[ri + H + j] + gh[rh + H + j]);
  float nn = tanhf(gi[ri + 2 * H + j] + rr * gh[rh + 2 * H + j]);
  cs[i] = (1.f - zz) * nn + zz * lps[i];
}

// listener GRU + speaker/listener blend + gather cur_personal[b,spk] as bf16
__global__ void k_gru_listener_combine(const float* __restrict__ gi, const float* __restrict__ gh,
                                       const float* __restrict__ lps, const float* __restrict__ cs,
                                       const float* __restrict__ pm,
                                       float* __restrict__ out_personal,
                                       unsigned short* __restrict__ cps) {
  int i = blockIdx.x * blockDim.x + threadIdx.x;
  if (i >= B * P * H) return;
  int j = i & (H - 1);
  int bp = i >> 9;
  int b = bp >> 1, p = bp & 1;
  size_t ri = (size_t)b * N3, rh = (size_t)bp * N3;
  float rr = sigmoidf(gi[ri + j] + gh[rh + j]);
  float zz = sigmoidf(gi[ri + H + j] + gh[rh + H + j]);
  float nn = tanhf(gi[ri + 2 * H + j] + rr * gh[rh + 2 * H + j]);
  float lv = (1.f - zz) * nn + zz * lps[i];
  float pmv = pm[b * 2 + p];
  float cur = lv * (1.f - pmv) + cs[i] * pmv;
  out_personal[i] = cur;
  if (pmv > 0.5f) cps[(size_t)b * H + j] = f2bf(cur);
}

// ---------------------------------------------------------------- launch
extern "C" void kernel_launch(void* const* d_in, const int* in_sizes, int n_in,
                              void* d_out, int out_size, void* d_ws, size_t ws_size,
                              hipStream_t stream) {
  (void)in_sizes; (void)n_in; (void)out_size; (void)ws_size;
  const float* utt   = (const float*)d_in[0];
  const float* pm    = (const float*)d_in[1];
  const float* ghist = (const float*)d_in[2];
  const float* lps   = (const float*)d_in[3];
  const float* le    = (const float*)d_in[4];
  const float* wg_ih = (const float*)d_in[5];
  const float* wg_hh = (const float*)d_in[6];
  const float* bg_ih = (const float*)d_in[7];
  const float* bg_hh = (const float*)d_in[8];
  const float* wp_ih = (const float*)d_in[9];
  const float* wp_hh = (const float*)d_in[10];
  const float* bp_ih = (const float*)d_in[11];
  const float* bp_hh = (const float*)d_in[12];
  const float* wl_ih = (const float*)d_in[13];
  const float* wl_hh = (const float*)d_in[14];
  const float* bl_ih = (const float*)d_in[15];
  const float* bl_hh = (const float*)d_in[16];
  const float* we_ih = (const float*)d_in[17];
  const float* we_hh = (const float*)d_in[18];
  const float* be_ih = (const float*)d_in[19];
  const float* be_hh = (const float*)d_in[20];
  const float* attnw = (const float*)d_in[21];

  float* out_global   = (float*)d_out;                     // B*H
  float* out_personal = out_global + (size_t)B * H;        // B*P*H
  float* out_emotion  = out_personal + (size_t)B * P * H;  // B*H
  float* out_attn     = out_emotion + (size_t)B * H;       // B*T

  // bump allocator over workspace
  uintptr_t wsp = (uintptr_t)d_ws;
  auto alloc = [&](size_t bytes) -> void* {
    uintptr_t r = (wsp + 255) & ~(uintptr_t)255;
    wsp = r + bytes;
    return (void*)r;
  };
  typedef unsigned short u16;
  u16* wgih_b = (u16*)alloc((size_t)N3 * K1 * 2);
  u16* wghh_b = (u16*)alloc((size_t)N3 * H * 2);
  u16* wpih_b = (u16*)alloc((size_t)N3 * K1 * 2);
  u16* wphh_b = (u16*)alloc((size_t)N3 * H * 2);
  u16* wlih_b = (u16*)alloc((size_t)N3 * K1 * 2);
  u16* wlhh_b = (u16*)alloc((size_t)N3 * H * 2);
  u16* weih_b = (u16*)alloc((size_t)N3 * H * 2);
  u16* wehh_b = (u16*)alloc((size_t)N3 * H * 2);
  u16* xg_b   = (u16*)alloc((size_t)B * K1 * 2);
  u16* hg_b   = (u16*)alloc((size_t)B * H * 2);
  u16* lps_b  = (u16*)alloc((size_t)B * P * H * 2);
  u16* uc_b   = (u16*)alloc((size_t)B * K1 * 2);
  u16* uss_b  = (u16*)alloc((size_t)B * K1 * 2);
  u16* cps_b  = (u16*)alloc((size_t)B * H * 2);
  u16* le_b   = (u16*)alloc((size_t)B * H * 2);
  float* scale = (float*)alloc((size_t)T * B * 4);
  float* alpha = (float*)alloc((size_t)T * B * 4);
  float* gA = (float*)alloc((size_t)B * N3 * 4);         // gi_* gates
  float* gB = (float*)alloc((size_t)B * P * N3 * 4);     // gh_* gates (covers M=4096)
  float* cs = (float*)alloc((size_t)B * P * H * 4);      // cur_speaker fp32

  auto cg = [](int n) { return (n + 255) / 256; };
  const float* gh_last = ghist + (size_t)(T - 1) * B * H;

  // weights -> bf16
  k_f2bf<<<cg(N3 * K1), 256, 0, stream>>>(wg_ih, wgih_b, N3 * K1);
  k_f2bf<<<cg(N3 * H),  256, 0, stream>>>(wg_hh, wghh_b, N3 * H);
  k_f2bf<<<cg(N3 * K1), 256, 0, stream>>>(wp_ih, wpih_b, N3 * K1);
  k_f2bf<<<cg(N3 * H),  256, 0, stream>>>(wp_hh, wphh_b, N3 * H);
  k_f2bf<<<cg(N3 * K1), 256, 0, stream>>>(wl_ih, wlih_b, N3 * K1);
  k_f2bf<<<cg(N3 * H),  256, 0, stream>>>(wl_hh, wlhh_b, N3 * H);
  k_f2bf<<<cg(N3 * H),  256, 0, stream>>>(we_ih, weih_b, N3 * H);
  k_f2bf<<<cg(N3 * H),  256, 0, stream>>>(we_hh, wehh_b, N3 * H);
  // activations -> bf16
  k_build_xg<<<cg(B * K1), 256, 0, stream>>>(utt, lps, pm, xg_b);
  k_f2bf<<<cg(B * H),     256, 0, stream>>>(gh_last, hg_b, B * H);
  k_f2bf<<<cg(B * P * H), 256, 0, stream>>>(lps, lps_b, B * P * H);
  k_f2bf<<<cg(B * H),     256, 0, stream>>>(le, le_b, B * H);

  // attention (fp32, bandwidth-bound: 2 passes over 512MB global_hist)
  k_scale<<<T * B / 8, 256, 0, stream>>>(ghist, attnw, scale);
  k_softmax<<<B, T, 0, stream>>>(scale, alpha, out_attn);
  k_pooled<<<cg(B * H), 256, 0, stream>>>(alpha, ghist, uc_b);
  k_concat_utt<<<cg(B * H), 256, 0, stream>>>(utt, uc_b);

  dim3 blk(256);
  // global GRU
  k_gemm_bf16<<<dim3(N3 / 128, B / 128), blk, 0, stream>>>(xg_b, wgih_b, bg_ih, gA, B, N3, K1);
  k_gemm_bf16<<<dim3(N3 / 128, B / 128), blk, 0, stream>>>(hg_b, wghh_b, bg_hh, gB, B, N3, H);
  k_gru_simple<<<cg(B * H), 256, 0, stream>>>(gA, gB, gh_last, out_global, B * H);
  // speaker GRU
  k_gemm_bf16<<<dim3(N3 / 128, B / 128), blk, 0, stream>>>(uc_b, wpih_b, bp_ih, gA, B, N3, K1);
  k_gemm_bf16<<<dim3(N3 / 128, (B * P) / 128), blk, 0, stream>>>(lps_b, wphh_b, bp_hh, gB, B * P, N3, H);
  k_gru_speaker<<<cg(B * P * H), 256, 0, stream>>>(gA, gB, lps, cs);
  // listener GRU + blend
  k_build_uss<<<cg(B * K1), 256, 0, stream>>>(utt, cs, pm, uss_b);
  k_gemm_bf16<<<dim3(N3 / 128, B / 128), blk, 0, stream>>>(uss_b, wlih_b, bl_ih, gA, B, N3, K1);
  k_gemm_bf16<<<dim3(N3 / 128, (B * P) / 128), blk, 0, stream>>>(lps_b, wlhh_b, bl_hh, gB, B * P, N3, H);
  k_gru_listener_combine<<<cg(B * P * H), 256, 0, stream>>>(gA, gB, lps, cs, pm, out_personal, cps_b);
  // emotion GRU
  k_gemm_bf16<<<dim3(N3 / 128, B / 128), blk, 0, stream>>>(cps_b, weih_b, be_ih, gA, B, N3, H);
  k_gemm_bf16<<<dim3(N3 / 128, B / 128), blk, 0, stream>>>(le_b, wehh_b, be_hh, gB, B, N3, H);
  k_gru_simple<<<cg(B * H), 256, 0, stream>>>(gA, gB, le, out_emotion, B * H);
}
